// GCN_75118978007273
// MI455X (gfx1250) — matrix operand order repaired and checked by reading, not compiled, not run on hardware
//
#include <hip/hip_runtime.h>
#include <hip/hip_bf16.h>

// ---------------------------------------------------------------------------
// GCN on MI455X (gfx1250, wave32).
//   cnt (int degree histogram) -> dinv = rsqrt(cnt+1) ; CSR-by-dst build
//   W1^T -> f16 ; Hw = x @ W1            [f16 WMMA, f32 accum, x streamed once]
//   Ha = relu(gather-agg(Hw) + b1)       [CSR gather, no float atomics]
//   W2^T -> f16 ; Hw = Ha @ W2
//   Ha = relu(gather-agg(Hw) + b2)
//   pre1/pre2 per-node dots with lin_W halves ; out[e] = pre1[s]+pre2[d]+lin_b
// ---------------------------------------------------------------------------

typedef _Float16 v16h __attribute__((ext_vector_type(16)));
typedef float    v8f  __attribute__((ext_vector_type(8)));

#define N_NODES 16384
#define HIDDEN  256
#define N_EDGES 1048576

#define TM 64
#define TN 256
#define TK 32
#define LDSPAD 8   // pad rows to 40 halves = 80B (16B multiple)

__device__ inline int atomAddI32(int* p, int v) {
  return __hip_atomic_fetch_add(p, v, __ATOMIC_RELAXED, __HIP_MEMORY_SCOPE_AGENT);
}

// ---------------------------------------------------------------------------
// degree histogram (int), dinv = rsqrt(cnt+1)  (+1 = self loop)
// ---------------------------------------------------------------------------
__global__ void gcn_cnt_zero(int* __restrict__ cnt, int n) {
  int i = blockIdx.x * blockDim.x + threadIdx.x;
  if (i < n) cnt[i] = 0;
}

__global__ void gcn_cnt_edges(const long long* __restrict__ eidx,
                              int* __restrict__ cnt, int E) {
  int e = blockIdx.x * blockDim.x + threadIdx.x;
  if (e < E) atomAddI32(&cnt[(int)eidx[(size_t)E + e]], 1);
}

__global__ void gcn_make_dinv(const int* __restrict__ cnt,
                              float* __restrict__ dinv, int n) {
  int i = blockIdx.x * blockDim.x + threadIdx.x;
  if (i < n) dinv[i] = rsqrtf((float)(cnt[i] + 1));
}

// ---------------------------------------------------------------------------
// single-block exclusive scan of 16384 ints -> rowstart[n+1], cursor copy
// 256 threads x 64-element serial chunks, LDS Hillis-Steele across threads
// ---------------------------------------------------------------------------
__global__ __launch_bounds__(256)
void gcn_scan(const int* __restrict__ cnt, int* __restrict__ rowstart,
              int* __restrict__ cursor, int n) {
  __shared__ int tsum[256];
  const int t = threadIdx.x;
  const int base = t * 64;
  int s = 0;
#pragma unroll 4
  for (int j = 0; j < 64; ++j) s += cnt[base + j];
  tsum[t] = s;
  __syncthreads();
  for (int off = 1; off < 256; off <<= 1) {
    int v = (t >= off) ? tsum[t - off] : 0;
    __syncthreads();
    tsum[t] += v;
    __syncthreads();
  }
  int run = (t == 0) ? 0 : tsum[t - 1];
  for (int j = 0; j < 64; ++j) {
    rowstart[base + j] = run;
    cursor[base + j]   = run;
    run += cnt[base + j];
  }
  if (t == 255) rowstart[n] = run;
}

// ---------------------------------------------------------------------------
// CSR fill: csr_src[pos], csr_w[pos] = src, dinv[src]*dinv[dst], grouped by dst
// ---------------------------------------------------------------------------
__global__ void gcn_csr_fill(const long long* __restrict__ eidx,
                             const float* __restrict__ dinv,
                             int* __restrict__ cursor,
                             int* __restrict__ csr_src,
                             float* __restrict__ csr_w, int E) {
  int e = blockIdx.x * blockDim.x + threadIdx.x;
  if (e >= E) return;
  int s = (int)eidx[e];
  int d = (int)eidx[(size_t)E + e];
  int pos = atomAddI32(&cursor[d], 1);
  csr_src[pos] = s;
  csr_w[pos]   = dinv[s] * dinv[d];
}

// ---------------------------------------------------------------------------
// Wt[n][k] = (f16) W[k][n]   (W: [K,N] f32 row-major -> Wt: [N,K] f16)
// ---------------------------------------------------------------------------
__global__ void gcn_transpose_f16(const float* __restrict__ W,
                                  _Float16* __restrict__ Wt, int K, int N) {
  size_t id = (size_t)blockIdx.x * blockDim.x + threadIdx.x; // over N*K, k fast
  if (id >= (size_t)N * K) return;
  int n = (int)(id / K);
  int k = (int)(id % K);
  Wt[id] = (_Float16)W[(size_t)k * N + n];
}

// ---------------------------------------------------------------------------
// C[M,N] = A[M,K](f32) @ Bt[N,K](f16)^T via v_wmma_f32_16x16x32_f16.
// Block: 256 threads = 8 waves; block tile 64x256 (full N per block so the
// 1 GiB `x` streams from HBM exactly once; Wt re-reads are L2-resident).
// Wave tile 32x64 -> 2x4 accumulators of 16x16.
// ---------------------------------------------------------------------------
__global__ __launch_bounds__(256)
void gcn_gemm_f16(const float* __restrict__ A, const _Float16* __restrict__ Bt,
                  float* __restrict__ C, int M, int N, int K) {
  __shared__ _Float16 As[TM][TK + LDSPAD];
  __shared__ _Float16 Bs[TN][TK + LDSPAD];

  const int tid  = threadIdx.x;
  const int lane = tid & 31;
  const int w    = tid >> 5;      // wave 0..7
  const int wm   = w & 1;         // 2 wave-rows  (32 M each)
  const int wn   = w >> 1;        // 4 wave-cols  (64 N each)
  const int half = lane >> 4;     // 0/1
  const int l16  = lane & 15;
  const int m0   = blockIdx.x * TM;

  v8f acc[2][4] = {};

  const int arow = tid >> 2;      // 0..63 : A fill row
  const int ac4  = tid & 3;       // A fill float4 col group
  const int brow = tid >> 2;      // 0..63 : B fill base row
  const int bk8  = tid & 3;       // B fill 8-half col group

  for (int k0 = 0; k0 < K; k0 += TK) {
    __syncthreads();
    // ---- fill As: 64x32 f32 -> f16 (2 float4 per thread) ----
    {
      const float* src = A + (size_t)(m0 + arow) * K + k0;
      if (k0 + TK < K) __builtin_prefetch(src + TK, 0, 0);  // global_prefetch_b8
#pragma unroll
      for (int i = 0; i < 2; ++i) {
        float4 v = *(const float4*)(src + (ac4 + 4 * i) * 4);
        int c = (ac4 + 4 * i) * 4;
        As[arow][c + 0] = (_Float16)v.x;
        As[arow][c + 1] = (_Float16)v.y;
        As[arow][c + 2] = (_Float16)v.z;
        As[arow][c + 3] = (_Float16)v.w;
      }
    }
    // ---- fill Bs: 256x32 f16 (4 x 16B per thread) ----
#pragma unroll
    for (int g = 0; g < 4; ++g) {
      int n = brow + 64 * g;
      uint4 v = *(const uint4*)(Bt + (size_t)n * K + k0 + bk8 * 8);
      *(uint4*)&Bs[n][bk8 * 8] = v;
    }
    __syncthreads();

    // ---- fragments ----
    // A: lane holds M=l16; lo 16B = K[half*8 .. +8), hi 16B = K[16+half*8 .. +8)
    v16h afrag[2];
#pragma unroll
    for (int i = 0; i < 2; ++i) {
      const _Float16* ap = &As[wm * 32 + i * 16 + l16][0];
      union { uint4 u[2]; v16h h; } t;
      t.u[0] = *(const uint4*)(ap + half * 8);
      t.u[1] = *(const uint4*)(ap + 16 + half * 8);
      afrag[i] = t.h;
    }
    // B: lane holds N=l16; lo 16B = K[half*8 .. +8), hi 16B = K[16+half*8 .. +8)
    v16h bfrag[4];
#pragma unroll
    for (int j = 0; j < 4; ++j) {
      const _Float16* bp = &Bs[wn * 64 + j * 16 + l16][0];
      union { uint4 u[2]; v16h h; } t;
      t.u[0] = *(const uint4*)(bp + half * 8);
      t.u[1] = *(const uint4*)(bp + 16 + half * 8);
      bfrag[j] = t.h;
    }
#pragma unroll
    for (int i = 0; i < 2; ++i)
#pragma unroll
      for (int j = 0; j < 4; ++j)
        acc[i][j] = __builtin_amdgcn_wmma_f32_16x16x32_f16(
            false, afrag[i], false, bfrag[j], (short)0, acc[i][j], false, false);
  }

  // ---- epilogue: VGPR r of a 16x16 f32 tile holds M=r+8*half, N=l16 ----
#pragma unroll
  for (int i = 0; i < 2; ++i)
#pragma unroll
    for (int j = 0; j < 4; ++j)
#pragma unroll
      for (int r = 0; r < 8; ++r) {
        int row = m0 + wm * 32 + i * 16 + half * 8 + r;
        int col = wn * 64 + j * 16 + l16;
        C[(size_t)row * N + col] = acc[i][j][r];
      }
}

// ---------------------------------------------------------------------------
// Hout[i] = relu( b + Hw[i]*dinv[i]^2 + sum_e Hw[csr_src[e]] * csr_w[e] )
// 64-thread group per node (float4 channels), 4 nodes per block.
// Pure gather: no float atomics; Hw is L2-resident (16 MB).
// ---------------------------------------------------------------------------
__global__ __launch_bounds__(256)
void gcn_aggregate(const float* __restrict__ Hw, const float* __restrict__ dinv,
                   const int* __restrict__ rowstart,
                   const int* __restrict__ csr_src,
                   const float* __restrict__ csr_w,
                   const float* __restrict__ bias,
                   float* __restrict__ Hout, int n) {
  const int tid  = threadIdx.x;
  const int node = blockIdx.x * 4 + (tid >> 6);
  const int c4   = tid & 63;
  const float4* hw = (const float4*)Hw;

  float di = dinv[node];
  float s2 = di * di;
  float4 acc = hw[(size_t)node * 64 + c4];
  acc.x *= s2; acc.y *= s2; acc.z *= s2; acc.w *= s2;

  const int beg = rowstart[node];
  const int end = rowstart[node + 1];
  for (int e = beg; e < end; ++e) {
    int   s = csr_src[e];
    float wgt = csr_w[e];
    float4 v = hw[(size_t)s * 64 + c4];
    acc.x += v.x * wgt; acc.y += v.y * wgt;
    acc.z += v.z * wgt; acc.w += v.w * wgt;
  }
  float4 bb = ((const float4*)bias)[c4];
  acc.x = fmaxf(acc.x + bb.x, 0.0f);
  acc.y = fmaxf(acc.y + bb.y, 0.0f);
  acc.z = fmaxf(acc.z + bb.z, 0.0f);
  acc.w = fmaxf(acc.w + bb.w, 0.0f);
  ((float4*)Hout)[(size_t)node * 64 + c4] = acc;
}

// ---------------------------------------------------------------------------
// pre1[i] = dot(H[i], linW[0:256]) ; pre2[i] = dot(H[i], linW[256:512])
// one wave32 per node, shfl_xor tree reduction
// ---------------------------------------------------------------------------
__global__ void gcn_node_score(const float* __restrict__ H,
                               const float* __restrict__ linW,
                               float* __restrict__ pre1,
                               float* __restrict__ pre2, int n) {
  int id = blockIdx.x * blockDim.x + threadIdx.x;   // over n*32
  int node = id >> 5;
  int l = id & 31;
  if (node >= n) return;
  const float4* h  = (const float4*)(H + (size_t)node * HIDDEN);
  const float4* w0 = (const float4*)linW;
  const float4* w1 = (const float4*)(linW + HIDDEN);
  float p1 = 0.0f, p2 = 0.0f;
#pragma unroll
  for (int j = 0; j < 2; ++j) {
    int c4 = l * 2 + j;                  // 0..63
    float4 a = h[c4], wa = w0[c4], wb = w1[c4];
    p1 += a.x * wa.x + a.y * wa.y + a.z * wa.z + a.w * wa.w;
    p2 += a.x * wb.x + a.y * wb.y + a.z * wb.z + a.w * wb.w;
  }
#pragma unroll
  for (int off = 16; off > 0; off >>= 1) {
    p1 += __shfl_xor(p1, off, 32);
    p2 += __shfl_xor(p2, off, 32);
  }
  if (l == 0) { pre1[node] = p1; pre2[node] = p2; }
}

// ---------------------------------------------------------------------------
// out[e] = pre1[src] + pre2[dst] + lin_b
// ---------------------------------------------------------------------------
__global__ void gcn_edge_out(const long long* __restrict__ eidx,
                             const float* __restrict__ pre1,
                             const float* __restrict__ pre2,
                             const float* __restrict__ linb,
                             float* __restrict__ out, int E) {
  int e = blockIdx.x * blockDim.x + threadIdx.x;
  if (e >= E) return;
  int s = (int)eidx[e];
  int d = (int)eidx[(size_t)E + e];
  out[e] = pre1[s] + pre2[d] + linb[0];
}

// ---------------------------------------------------------------------------
extern "C" void kernel_launch(void* const* d_in, const int* in_sizes, int n_in,
                              void* d_out, int out_size, void* d_ws, size_t ws_size,
                              hipStream_t stream) {
  const float*     x    = (const float*)d_in[0];
  const long long* eidx = (const long long*)d_in[1];   // int64 per reference
  const float*     W1   = (const float*)d_in[2];
  const float*     b1   = (const float*)d_in[3];
  const float*     W2   = (const float*)d_in[4];
  const float*     b2   = (const float*)d_in[5];
  const float*     linW = (const float*)d_in[6];
  const float*     linb = (const float*)d_in[7];
  float*           out  = (float*)d_out;

  const int n = N_NODES, E = N_EDGES;

  // workspace layout (bytes)
  char* ws = (char*)d_ws;
  float*    dinv     = (float*)(ws + (size_t)0);
  float*    pre1     = (float*)(ws + ((size_t)64 << 10));
  float*    pre2     = (float*)(ws + ((size_t)128 << 10));
  int*      cnt      = (int*)  (ws + ((size_t)192 << 10));
  int*      rowstart = (int*)  (ws + ((size_t)256 << 10));   // n+1 ints
  int*      cursor   = (int*)  (ws + ((size_t)384 << 10));
  _Float16* Wt       = (_Float16*)(ws + ((size_t)1 << 20));  //  8 MB
  int*      csr_src  = (int*)  (ws + ((size_t)9 << 20));     //  4 MB
  float*    csr_w    = (float*)(ws + ((size_t)13 << 20));    //  4 MB
  float*    Hw       = (float*)(ws + ((size_t)17 << 20));    // 16 MB
  float*    Ha       = (float*)(ws + ((size_t)33 << 20));    // 16 MB

  const int nt = 256;

  // ---- degrees, dinv, CSR-by-dst ----
  gcn_cnt_zero<<<n / nt, nt, 0, stream>>>(cnt, n);
  gcn_cnt_edges<<<E / nt, nt, 0, stream>>>(eidx, cnt, E);
  gcn_make_dinv<<<n / nt, nt, 0, stream>>>(cnt, dinv, n);
  gcn_scan<<<1, nt, 0, stream>>>(cnt, rowstart, cursor, n);
  gcn_csr_fill<<<E / nt, nt, 0, stream>>>(eidx, dinv, cursor, csr_src, csr_w, E);

  // ---- layer 1 ----
  gcn_transpose_f16<<<(HIDDEN * N_NODES) / nt, nt, 0, stream>>>(W1, Wt, N_NODES, HIDDEN);
  gcn_gemm_f16<<<dim3(n / TM), nt, 0, stream>>>(x, Wt, Hw, n, HIDDEN, N_NODES);
  gcn_aggregate<<<n / 4, nt, 0, stream>>>(Hw, dinv, rowstart, csr_src, csr_w, b1, Ha, n);

  // ---- layer 2 ----
  gcn_transpose_f16<<<(HIDDEN * HIDDEN) / nt, nt, 0, stream>>>(W2, Wt, HIDDEN, HIDDEN);
  gcn_gemm_f16<<<dim3(n / TM), nt, 0, stream>>>(Ha, Wt, Hw, n, HIDDEN, HIDDEN);
  gcn_aggregate<<<n / 4, nt, 0, stream>>>(Hw, dinv, rowstart, csr_src, csr_w, b2, Ha, n);

  // ---- edge scoring: per-node partial dots, then trivial per-edge combine ----
  gcn_node_score<<<(n * 32) / nt, nt, 0, stream>>>(Ha, linW, pre1, pre2, n);
  gcn_edge_out<<<E / nt, nt, 0, stream>>>(eidx, pre1, pre2, linb, out, E);
}
